// TransactionGNN_25589415150280
// MI455X (gfx1250) — matrix operand order repaired and checked by reading, not compiled, not run on hardware
//
#include <hip/hip_runtime.h>
#include <hip/hip_bf16.h>
#include <math.h>

#define N_NODES 100000
#define IN_CH   64
#define HID_CH  128
#define N_EDGES 1600000

typedef __attribute__((ext_vector_type(16))) __bf16 v16bf;
typedef __attribute__((ext_vector_type(8)))  float  v8f;

// ---------------- degree / normalization ----------------

__global__ void init_deg_kernel(float* deg) {
    int i = blockIdx.x * blockDim.x + threadIdx.x;
    if (i < N_NODES) deg[i] = 1.0f;  // self-loop contributes 1 to every node
}

__global__ void count_deg_kernel(const long long* __restrict__ dst, float* deg) {
    int e = blockIdx.x * blockDim.x + threadIdx.x;
    if (e < N_EDGES) atomicAdd(&deg[(int)dst[e]], 1.0f);
}

__global__ void rsqrt_kernel(float* deg) {
    int i = blockIdx.x * blockDim.x + threadIdx.x;
    if (i < N_NODES) deg[i] = rsqrtf(deg[i]);  // deg >= 1 always
}

__global__ void zero_f32_kernel(float* p, long long n) {
    long long i = (long long)blockIdx.x * blockDim.x + threadIdx.x;
    if (i < n) p[i] = 0.0f;
}

// ---------------- layer-1 dense transform: h = x @ W1^T  (bf16 WMMA) ------
// Block = 256 threads = 8 waves. Wave w computes the 16x16 tile at
// rows [blockIdx.x*16, +16), cols [w*16, +16).  K = 64 -> 2x wmma K=32.

__global__ __launch_bounds__(256)
void gemm1_wmma_kernel(const float* __restrict__ x,
                       const float* __restrict__ W1,
                       __bf16* __restrict__ h) {
    const int wave = threadIdx.x >> 5;   // 0..7 -> output column tile
    const int lane = threadIdx.x & 31;
    const int half = lane >> 4;          // 0 or 1
    const int l16  = lane & 15;
    const int row0 = blockIdx.x * 16;    // 100000 == 6250*16, always in-bounds
    const int n0   = wave * 16;

    v8f c = {};
    #pragma unroll
    for (int kc = 0; kc < 2; ++kc) {
        const int kb = kc * 32;
        // A fragment: 16x32 bf16.  lane(l16,half) row M=l16,
        //   elems 0..7  -> K = kb + half*16 + e
        //   elems 8..15 -> K = kb + half*16 + 8 + e
        const float* xr = x + (size_t)(row0 + l16) * IN_CH + kb + half * 16;
        v16bf a;
        #pragma unroll
        for (int e = 0; e < 8; ++e) {
            a[e]     = (__bf16)xr[e];
            a[e + 8] = (__bf16)xr[8 + e];
        }
        // B fragment: 32x16 bf16, B[k][n] = W1[n0+n][k].
        //   lane: N = l16, elems e -> K = kb + half*16 + e
        const float* wr = W1 + (size_t)(n0 + l16) * IN_CH + kb + half * 16;
        v16bf b;
        #pragma unroll
        for (int e = 0; e < 16; ++e) b[e] = (__bf16)wr[e];

        c = __builtin_amdgcn_wmma_f32_16x16x32_bf16(
                /*neg_a=*/false, a, /*neg_b=*/false, b,
                /*c_mod=*/(short)0, c, /*reuse_a=*/false, /*reuse_b=*/false);
    }
    // D layout: VGPR r -> M = r + half*8, N = l16
    #pragma unroll
    for (int r = 0; r < 8; ++r) {
        const int row = row0 + r + half * 8;
        const int col = n0 + l16;
        h[(size_t)row * HID_CH + col] = (__bf16)c[r];
    }
}

// ---------------- layer-1 edge aggregation (bf16 gather, f32 atomic) ------
// one thread = (edge, 4 channels); 32 threads cover one edge's 128 channels.

__global__ void scatter1_kernel(const long long* __restrict__ src,
                                const long long* __restrict__ dst,
                                const float* __restrict__ dis,
                                const __bf16* __restrict__ h,
                                float* __restrict__ agg) {
    long long idx = (long long)blockIdx.x * blockDim.x + threadIdx.x;
    if (idx >= (long long)N_EDGES * 32) return;
    const int e  = (int)(idx >> 5);
    const int c0 = ((int)idx & 31) * 4;
    const int s  = (int)src[e];
    const int d  = (int)dst[e];
    const float norm = dis[s] * dis[d];
    const __bf16* hp = h + (size_t)s * HID_CH + c0;
    float* ap = agg + (size_t)d * HID_CH + c0;
    #pragma unroll
    for (int j = 0; j < 4; ++j)
        atomicAdd(ap + j, (float)hp[j] * norm);
}

// self-loop term + bias + ReLU; writes activated h1 in place (bf16)
__global__ void epilogue1_kernel(const float* __restrict__ agg,
                                 __bf16* __restrict__ h,
                                 const float* __restrict__ dis,
                                 const float* __restrict__ b1) {
    long long i = (long long)blockIdx.x * blockDim.x + threadIdx.x;
    if (i >= (long long)N_NODES * HID_CH) return;
    const int node = (int)(i >> 7);
    const int c    = (int)(i & (HID_CH - 1));
    const float self = dis[node] * dis[node];
    float v = agg[i] + self * (float)h[i] + b1[c];
    h[i] = (__bf16)fmaxf(v, 0.0f);
}

// ---------------- layer 2: h2 = h1 @ W2^T  (per-node dot, 128 wide) ------

__global__ void gemm2_kernel(const __bf16* __restrict__ h1,
                             const float* __restrict__ W2,
                             float* __restrict__ h2) {
    int i = blockIdx.x * blockDim.x + threadIdx.x;
    if (i >= N_NODES) return;
    const __bf16* hp = h1 + (size_t)i * HID_CH;
    float acc = 0.0f;
    #pragma unroll 8
    for (int c = 0; c < HID_CH; ++c) acc += (float)hp[c] * W2[c];
    h2[i] = acc;
}

__global__ void scatter2_kernel(const long long* __restrict__ src,
                                const long long* __restrict__ dst,
                                const float* __restrict__ dis,
                                const float* __restrict__ h2,
                                float* __restrict__ agg2) {
    int e = blockIdx.x * blockDim.x + threadIdx.x;
    if (e >= N_EDGES) return;
    const int s = (int)src[e];
    const int d = (int)dst[e];
    atomicAdd(&agg2[d], h2[s] * dis[s] * dis[d]);
}

__global__ void final_kernel(const float* __restrict__ agg2,
                             const float* __restrict__ h2,
                             const float* __restrict__ dis,
                             const float* __restrict__ b2,
                             float* __restrict__ out) {
    int i = blockIdx.x * blockDim.x + threadIdx.x;
    if (i >= N_NODES) return;
    float v = agg2[i] + dis[i] * dis[i] * h2[i] + b2[0];
    out[i] = 1.0f / (1.0f + expf(-v));
}

// ---------------- host launcher ----------------

static inline size_t align_up(size_t x, size_t a) { return (x + a - 1) & ~(a - 1); }

extern "C" void kernel_launch(void* const* d_in, const int* in_sizes, int n_in,
                              void* d_out, int out_size, void* d_ws, size_t ws_size,
                              hipStream_t stream) {
    (void)in_sizes; (void)n_in; (void)out_size; (void)ws_size;

    const float*     x  = (const float*)d_in[0];                  // [N, 64]
    const long long* ei = (const long long*)d_in[1];              // [2, E] int64
    const float*     W1 = (const float*)d_in[2];                  // [128, 64]
    const float*     b1 = (const float*)d_in[3];                  // [128]
    const float*     W2 = (const float*)d_in[4];                  // [1, 128]
    const float*     b2 = (const float*)d_in[5];                  // [1]
    float*           out = (float*)d_out;                         // [N, 1]

    const long long* src = ei;
    const long long* dst = ei + N_EDGES;

    char* ws = (char*)d_ws;
    size_t o = 0;
    float*  dis  = (float*)(ws + o);  o = align_up(o + (size_t)N_NODES * 4, 256);
    __bf16* h    = (__bf16*)(ws + o); o = align_up(o + (size_t)N_NODES * HID_CH * 2, 256);
    float*  agg1 = (float*)(ws + o);  o = align_up(o + (size_t)N_NODES * HID_CH * 4, 256);
    float*  h2   = (float*)(ws + o);  o = align_up(o + (size_t)N_NODES * 4, 256);
    float*  agg2 = (float*)(ws + o);  o = align_up(o + (size_t)N_NODES * 4, 256);

    const int B = 256;
    const int gN  = (N_NODES + B - 1) / B;
    const int gE  = (N_EDGES + B - 1) / B;

    // normalization: deg (with self-loops) -> rsqrt
    init_deg_kernel<<<gN, B, 0, stream>>>(dis);
    count_deg_kernel<<<gE, B, 0, stream>>>(dst, dis);
    rsqrt_kernel<<<gN, B, 0, stream>>>(dis);

    // layer 1: dense transform via bf16 WMMA (6250 blocks x 8 waves)
    gemm1_wmma_kernel<<<N_NODES / 16, 256, 0, stream>>>(x, W1, h);

    // layer 1: edge aggregation
    const long long nAgg1 = (long long)N_NODES * HID_CH;
    zero_f32_kernel<<<(unsigned)((nAgg1 + B - 1) / B), B, 0, stream>>>(agg1, nAgg1);
    const long long nSc1 = (long long)N_EDGES * 32;
    scatter1_kernel<<<(unsigned)((nSc1 + B - 1) / B), B, 0, stream>>>(src, dst, dis, h, agg1);
    epilogue1_kernel<<<(unsigned)((nAgg1 + B - 1) / B), B, 0, stream>>>(agg1, h, dis, b1);

    // layer 2
    gemm2_kernel<<<gN, B, 0, stream>>>(h, W2, h2);
    zero_f32_kernel<<<gN, B, 0, stream>>>(agg2, (long long)N_NODES);
    scatter2_kernel<<<gE, B, 0, stream>>>(src, dst, dis, h2, agg2);
    final_kernel<<<gN, B, 0, stream>>>(agg2, h2, dis, b2, out);
}
